// TwoTwoTwoMultitaskAutoencoder_38929583571517
// MI455X (gfx1250) — matrix operand order repaired and checked by reading, not compiled, not run on hardware
//
#include <hip/hip_runtime.h>

// ---------------- problem constants ----------------
#define BB 32768
#define LL 512
#define HH 1024
#define EE 128
#define KK 8

typedef __attribute__((ext_vector_type(16))) _Float16 v16h;
typedef __attribute__((ext_vector_type(8)))  _Float16 v8h;
typedef __attribute__((ext_vector_type(8)))  float    v8f;

// Load a 16x32 f16 A-fragment chunk for one lane.
// Layout (16-bit A, 16x32): lane holds row M=lane&15; lanes<16 hold K in
// [k0, k0+16); lanes>=16 hold [k0+16, k0+32) (contiguous, matching B).
// Caller passes p = row_base + k0 + (lane>>4)*16 ; we read 8 halfs at p and 8 at p+8.
__device__ __forceinline__ v16h loadA16(const _Float16* p) {
  v8h lo = *(const v8h*)(p);
  v8h hi = *(const v8h*)(p + 8);
  v16h a;
#pragma unroll
  for (int i = 0; i < 8; ++i) { a[i] = lo[i]; a[i + 8] = hi[i]; }
  return a;
}

__device__ __forceinline__ float leaky(float v) { return v > 0.0f ? v : 0.01f * v; }

// ---------------- weight convert + transpose (fp32 KxN -> f16 NxK, batched) ----
__global__ void k_wT(const float* __restrict__ src, _Float16* __restrict__ dst,
                     int K, int N, int total) {
  int idx = blockIdx.x * 256 + threadIdx.x;
  if (idx >= total) return;
  int kn  = K * N;
  int b   = idx / kn;
  int rem = idx - b * kn;
  int k   = rem / N;
  int n   = rem - k * N;
  dst[(size_t)b * kn + (size_t)n * K + k] = (_Float16)src[idx];
}

// ---------------- stable counting sort by expert id --------------------------
__global__ void k_ids(const float* __restrict__ x, int* __restrict__ expOrig,
                      int* __restrict__ blkCnt) {
  __shared__ int cnt[KK];
  int t = threadIdx.x;
  if (t < KK) cnt[t] = 0;
  __syncthreads();
  int row = blockIdx.x * 256 + t;
  int e = (int)x[(size_t)row * (LL + 1) + LL];
  expOrig[row] = e;
  atomicAdd(&cnt[e], 1);
  __syncthreads();
  if (t < KK) blkCnt[blockIdx.x * KK + t] = cnt[t];
}

__global__ void k_scan(const int* __restrict__ blkCnt, int* __restrict__ blkOff) {
  // single-thread scan over 128 blocks x 8 keys (tiny)
  int totals[KK];
  for (int k = 0; k < KK; ++k) totals[k] = 0;
  for (int b = 0; b < 128; ++b)
    for (int k = 0; k < KK; ++k) totals[k] += blkCnt[b * KK + k];
  int run[KK];
  int s = 0;
  for (int k = 0; k < KK; ++k) { run[k] = s; s += totals[k]; }
  for (int b = 0; b < 128; ++b)
    for (int k = 0; k < KK; ++k) { blkOff[b * KK + k] = run[k]; run[k] += blkCnt[b * KK + k]; }
}

__global__ void k_scatter(const int* __restrict__ expOrig, const int* __restrict__ blkOff,
                          int* __restrict__ order, int* __restrict__ expSorted) {
  __shared__ int keys[256];
  int t = threadIdx.x;
  int row = blockIdx.x * 256 + t;
  int e = expOrig[row];
  keys[t] = e;
  __syncthreads();
  int rank = 0;
  for (int i = 0; i < 256; ++i) {           // stable in-block rank
    if (i >= t) break;
    rank += (keys[i] == e);
  }
  int dst = blkOff[blockIdx.x * KK + e] + rank;
  order[dst] = row;
  expSorted[dst] = e;
}

// ---------------- gather sorted rows of x into f16 (drop id column) ----------
__global__ void k_gather(const float* __restrict__ x, const int* __restrict__ order,
                         _Float16* __restrict__ Xs) {
  size_t idx = (size_t)blockIdx.x * 256 + threadIdx.x;   // B*L threads
  int s = (int)(idx >> 9);
  int c = (int)(idx & (LL - 1));
  Xs[idx] = (_Float16)x[(size_t)order[s] * (LL + 1) + c];
}

// ---------------- fused MoE autoencoder, 16 sorted rows per workgroup --------
#define HS 1032   // bufH row stride in halfs (16B-aligned, bank-skewed)
#define ZS 136    // bufZ row stride in halfs

__global__ __launch_bounds__(256) void fused_moe(
    const _Float16* __restrict__ Xs,
    const _Float16* __restrict__ We1T,   // 1024 x 512   (NxK)
    const _Float16* __restrict__ We2T,   // 128  x 1024
    const _Float16* __restrict__ Wd1T,   // 8 x 1024 x 128
    const _Float16* __restrict__ Wd2T,   // 8 x 512  x 1024
    const float* __restrict__ be1, const float* __restrict__ be2,
    const float* __restrict__ bd1, const float* __restrict__ bd2,
    const int* __restrict__ expSorted,
    float* __restrict__ out) {
  __shared__ _Float16 bufH[16 * HS];   // H1, then reused for HD
  __shared__ _Float16 bufZ[16 * ZS];

  const int tid  = threadIdx.x;
  const int wave = tid >> 5;
  const int lane = tid & 31;
  const int l15  = lane & 15;
  const int lhi  = lane >> 4;       // 0/1
  const int koff = lhi * 16;        // A-fragment K sub-offset
  const int kb   = lhi * 16;        // B-fragment K sub-offset
  const int tileBase = blockIdx.x * 16;

  // experts of the 8 output rows this lane's C fragment covers
  int rowExp[8];
#pragma unroll
  for (int r = 0; r < 8; ++r) rowExp[r] = expSorted[tileBase + r + 8 * lhi];

  // ================= GEMM1: H1 = leaky(Xs @ We1 + be1)  M=16 N=1024 K=512 ====
  const _Float16* aBase1 = Xs + (size_t)(tileBase + l15) * LL;
  for (int j = 0; j < 8; ++j) {
    const int n0 = (wave + 8 * j) * 16;
    v8f acc = {};
    const _Float16* bBase = We1T + (size_t)(n0 + l15) * LL;
#pragma unroll 4
    for (int kt = 0; kt < LL / 32; ++kt) {
      const int k0 = kt * 32;
      v16h a = loadA16(aBase1 + k0 + koff);
      v16h b = *(const v16h*)(bBase + k0 + kb);
      acc = __builtin_amdgcn_wmma_f32_16x16x32_f16(false, a, false, b, (short)0,
                                                   acc, false, false);
    }
    const float bias = be1[n0 + l15];
#pragma unroll
    for (int r = 0; r < 8; ++r) {
      float v = leaky(acc[r] + bias);
      bufH[(r + 8 * lhi) * HS + n0 + l15] = (_Float16)v;
    }
  }
  __syncthreads();

  // ================= GEMM2: Z = leaky(H1 @ We2 + be2)  M=16 N=128 K=1024 =====
  {
    const int n0 = wave * 16;
    v8f acc = {};
    const _Float16* bBase = We2T + (size_t)(n0 + l15) * HH;
    const _Float16* aB = bufH + l15 * HS;
#pragma unroll 4
    for (int kt = 0; kt < HH / 32; ++kt) {
      const int k0 = kt * 32;
      v16h a = loadA16(aB + k0 + koff);
      v16h b = *(const v16h*)(bBase + k0 + kb);
      acc = __builtin_amdgcn_wmma_f32_16x16x32_f16(false, a, false, b, (short)0,
                                                   acc, false, false);
    }
    const float bias = be2[n0 + l15];
#pragma unroll
    for (int r = 0; r < 8; ++r) {
      float v = leaky(acc[r] + bias);
      bufZ[(r + 8 * lhi) * ZS + n0 + l15] = (_Float16)v;
    }
  }
  __syncthreads();

  // ================= per-expert decoder (tile is sorted: usually 1 expert) ===
  const int eFirst = expSorted[tileBase];
  const int eLast  = expSorted[tileBase + 15];
  for (int e = eFirst; e <= eLast; ++e) {
    // GEMM3: HD = leaky(Z @ Wd1[e] + bd1[e])  M=16 N=1024 K=128
    for (int j = 0; j < 8; ++j) {
      const int n0 = (wave + 8 * j) * 16;
      v8f acc = {};
      const _Float16* bBase = Wd1T + (size_t)e * HH * EE + (size_t)(n0 + l15) * EE;
      const _Float16* aB = bufZ + l15 * ZS;
#pragma unroll
      for (int kt = 0; kt < EE / 32; ++kt) {
        const int k0 = kt * 32;
        v16h a = loadA16(aB + k0 + koff);
        v16h b = *(const v16h*)(bBase + k0 + kb);
        acc = __builtin_amdgcn_wmma_f32_16x16x32_f16(false, a, false, b, (short)0,
                                                     acc, false, false);
      }
      const float bias = bd1[e * HH + n0 + l15];
#pragma unroll
      for (int r = 0; r < 8; ++r) {
        float v = leaky(acc[r] + bias);
        if (rowExp[r] == e) bufH[(r + 8 * lhi) * HS + n0 + l15] = (_Float16)v;
      }
    }
    __syncthreads();

    // GEMM4: Y = HD @ Wd2[e] + bd2[e]  M=16 N=512 K=1024  (output in sorted order)
    for (int j = 0; j < 4; ++j) {
      const int n0 = (wave + 8 * j) * 16;
      v8f acc = {};
      const _Float16* bBase = Wd2T + (size_t)e * LL * HH + (size_t)(n0 + l15) * HH;
      const _Float16* aB = bufH + l15 * HS;
#pragma unroll 4
      for (int kt = 0; kt < HH / 32; ++kt) {
        const int k0 = kt * 32;
        v16h a = loadA16(aB + k0 + koff);
        v16h b = *(const v16h*)(bBase + k0 + kb);
        acc = __builtin_amdgcn_wmma_f32_16x16x32_f16(false, a, false, b, (short)0,
                                                     acc, false, false);
      }
      const float bias = bd2[e * LL + n0 + l15];
#pragma unroll
      for (int r = 0; r < 8; ++r) {
        if (rowExp[r] == e)
          out[(size_t)(tileBase + r + 8 * lhi) * LL + n0 + l15] = acc[r] + bias;
      }
    }
    __syncthreads();
  }
}

// ---------------- host-side orchestration ------------------------------------
extern "C" void kernel_launch(void* const* d_in, const int* in_sizes, int n_in,
                              void* d_out, int out_size, void* d_ws, size_t ws_size,
                              hipStream_t stream) {
  const float* x   = (const float*)d_in[0];
  const float* We1 = (const float*)d_in[1];
  const float* be1 = (const float*)d_in[2];
  const float* We2 = (const float*)d_in[3];
  const float* be2 = (const float*)d_in[4];
  const float* Wd1 = (const float*)d_in[5];
  const float* bd1 = (const float*)d_in[6];
  const float* Wd2 = (const float*)d_in[7];
  const float* bd2 = (const float*)d_in[8];
  float* out = (float*)d_out;

  char* ws = (char*)d_ws;
  size_t off = 0;
  auto alloc = [&](size_t bytes) -> void* {
    off = (off + 255) & ~(size_t)255;
    void* p = ws + off;
    off += bytes;
    return p;
  };

  _Float16* Xs   = (_Float16*)alloc((size_t)BB * LL * 2);
  _Float16* We1T = (_Float16*)alloc((size_t)LL * HH * 2);
  _Float16* We2T = (_Float16*)alloc((size_t)HH * EE * 2);
  _Float16* Wd1T = (_Float16*)alloc((size_t)KK * EE * HH * 2);
  _Float16* Wd2T = (_Float16*)alloc((size_t)KK * HH * LL * 2);
  int* order     = (int*)alloc((size_t)BB * 4);
  int* expSorted = (int*)alloc((size_t)BB * 4);
  int* expOrig   = (int*)alloc((size_t)BB * 4);
  int* blkCnt    = (int*)alloc(128 * KK * 4);
  int* blkOff    = (int*)alloc(128 * KK * 4);

  // weights -> f16, transposed to NxK
  k_wT<<<(LL * HH + 255) / 256, 256, 0, stream>>>(We1, We1T, LL, HH, LL * HH);
  k_wT<<<(HH * EE + 255) / 256, 256, 0, stream>>>(We2, We2T, HH, EE, HH * EE);
  k_wT<<<(KK * EE * HH + 255) / 256, 256, 0, stream>>>(Wd1, Wd1T, EE, HH, KK * EE * HH);
  k_wT<<<(KK * HH * LL + 255) / 256, 256, 0, stream>>>(Wd2, Wd2T, HH, LL, KK * HH * LL);

  // stable counting sort by expert id
  k_ids<<<BB / 256, 256, 0, stream>>>(x, expOrig, blkCnt);
  k_scan<<<1, 1, 0, stream>>>(blkCnt, blkOff);
  k_scatter<<<BB / 256, 256, 0, stream>>>(expOrig, blkOff, order, expSorted);

  // gather sorted rows as f16
  k_gather<<<(BB * LL) / 256, 256, 0, stream>>>(x, order, Xs);

  // fused WMMA pipeline
  fused_moe<<<BB / 16, 256, 0, stream>>>(Xs, We1T, We2T, Wd1T, Wd2T,
                                         be1, be2, bd1, bd2, expSorted, out);
}
